// EdgeGNN_13305808683494
// MI455X (gfx1250) — compile-verified, matrix operand order checked by reading, not run on hardware
//
#include <hip/hip_runtime.h>
#include <hip/hip_bf16.h>

#define BN_EPS 1e-5f

typedef __attribute__((ext_vector_type(16))) __bf16 v16bf;
typedef __attribute__((ext_vector_type(8)))  __bf16 v8bf;
typedef __attribute__((ext_vector_type(8)))  float  v8f;

union FragU { v16bf v; v8bf h[2]; };

// Fragment loader for CDNA5 16-bit WMMA operand layout (16x32 MxK / 32x16 KxN
// with pre-transposed B). Lane L: elements 0..7 hold k = kbase..kbase+7,
// elements 8..15 hold k = kbase+16..kbase+23, where kbase = ks*32 + (L>>4)*8.
__device__ inline v16bf load_frag16(const __bf16* __restrict__ p, int kbase) {
    FragU f;
    f.h[0] = *(const v8bf*)(p + kbase);
    f.h[1] = *(const v8bf*)(p + kbase + 16);
    return f.v;
}

// ---------------------------------------------------------------------------
// Tiled bf16 WMMA GEMM: C[M x NCOLS] = A[M x K] @ B[K x NCOLS]
//   Bt is B transposed to [NCOLS x K] bf16 (k-contiguous rows).
//   Block = 128 threads = 4 waves; each wave: 16 rows x NCOLS.
//   EPI 0: outF = acc                                              (GCN hW)
//   EPI 1: y = acc + shift[c] + addend[r,c]; outF=y; outB=bf16(y)  (skip)
//   EPI 2: y = relu(acc*scale[c] + shift[c]); outB=bf16(y)         (MLP bn)
//   GATHER: A row r = concat(hbf[sIdx[r]], hbf[dIdx[r]]) (K=256 concat)
// ---------------------------------------------------------------------------
template<int K, int NCOLS, int EPI, bool GATHER>
__global__ void __launch_bounds__(128)
gemm_wmma(const __bf16* __restrict__ A, const __bf16* __restrict__ Bt,
          const float* __restrict__ scale, const float* __restrict__ shift,
          const float* __restrict__ addend, float* __restrict__ outF,
          __bf16* __restrict__ outB, int M,
          const int* __restrict__ sIdx, const int* __restrict__ dIdx,
          const __bf16* __restrict__ hbf)
{
    constexpr int NT = NCOLS / 16;
    const int lane = threadIdx.x & 31;
    const int wave = threadIdx.x >> 5;
    const int r16  = lane & 15;
    const int kh   = lane >> 4;
    const int row0 = (blockIdx.x * 4 + wave) * 16;

    int arow = row0 + r16;
    if (arow >= M) arow = M - 1;   // clamp: keep EXEC all-ones for WMMA

    v8f acc[NT];
#pragma unroll
    for (int t = 0; t < NT; ++t)
#pragma unroll
        for (int j = 0; j < 8; ++j) acc[t][j] = 0.0f;

    const __bf16* Arow = nullptr;
    int sI = 0, dI = 0;
    if constexpr (GATHER) { sI = sIdx[arow]; dI = dIdx[arow]; }
    else                  { Arow = A + (long)arow * K; }

#pragma unroll
    for (int ks = 0; ks < K / 32; ++ks) {
        const int kb = ks * 32 + kh * 8;
        v16bf a;
        if constexpr (GATHER) {
            // each 8-wide span stays inside one 128-wide half of the concat
            FragU f;
            const int k0 = kb, k1 = kb + 16;
            const int n0 = (k0 < 128) ? sI : dI;
            const int n1 = (k1 < 128) ? sI : dI;
            f.h[0] = *(const v8bf*)(hbf + (long)n0 * 128 + (k0 & 127));
            f.h[1] = *(const v8bf*)(hbf + (long)n1 * 128 + (k1 & 127));
            a = f.v;
        } else {
            a = load_frag16(Arow, kb);
        }
#pragma unroll
        for (int nt = 0; nt < NT; ++nt) {
            const int col = nt * 16 + r16;
            v16bf b = load_frag16(Bt + (long)col * K, kb);
            acc[nt] = __builtin_amdgcn_wmma_f32_16x16x32_bf16(
                false, a, false, b, (short)0, acc[nt], false, false);
        }
    }

    // Epilogue. C/D layout: lane L, VGPR j -> row = j + 8*(L>>4), col = nt*16 + (L&15)
#pragma unroll
    for (int nt = 0; nt < NT; ++nt) {
        const int c = nt * 16 + r16;
#pragma unroll
        for (int j = 0; j < 8; ++j) {
            const int r = row0 + j + 8 * kh;
            if (r >= M) continue;
            const long o = (long)r * NCOLS + c;
            float v = acc[nt][j];
            if constexpr (EPI == 0) {
                outF[o] = v;
            } else if constexpr (EPI == 1) {
                float y = v + shift[c] + addend[o];
                outF[o] = y;
                outB[o] = (__bf16)y;
            } else {
                float y = v * scale[c] + shift[c];
                y = y > 0.0f ? y : 0.0f;
                outB[o] = (__bf16)y;
            }
        }
    }
}

// ------------------------- small utility kernels ---------------------------

__global__ void k_f32_to_bf16(const float* __restrict__ s, __bf16* __restrict__ d, long n) {
    long i = (long)blockIdx.x * blockDim.x + threadIdx.x;
    if (i < n) d[i] = (__bf16)s[i];
}

__global__ void k_set_i32(int* __restrict__ p, int v, int n) {
    int i = blockIdx.x * blockDim.x + threadIdx.x;
    if (i < n) p[i] = v;
}

// in-degree histogram over dst
__global__ void k_hist(const int* __restrict__ dst, int* __restrict__ cnt, int E) {
    int i = blockIdx.x * blockDim.x + threadIdx.x;
    if (i < E) atomicAdd(&cnt[dst[i]], 1);
}

// dinv = rsqrt(in_degree + self_loop)
__global__ void k_dinv(const int* __restrict__ cnt, float* __restrict__ dinv, int n) {
    int i = blockIdx.x * blockDim.x + threadIdx.x;
    if (i < n) dinv[i] = rsqrtf((float)cnt[i] + 1.0f);
}

// --- exclusive scan of cnt -> rowptr (CSR offsets), 1024-wide block scan ---
__global__ void k_scan_block(const int* __restrict__ cnt, int* __restrict__ excl,
                             int* __restrict__ bsum, int n) {
    __shared__ int sm[1024];
    const int i = blockIdx.x * 1024 + threadIdx.x;
    const int v = (i < n) ? cnt[i] : 0;
    sm[threadIdx.x] = v;
    __syncthreads();
    for (int ofs = 1; ofs < 1024; ofs <<= 1) {     // Hillis-Steele inclusive
        int t = (threadIdx.x >= ofs) ? sm[threadIdx.x - ofs] : 0;
        __syncthreads();
        sm[threadIdx.x] += t;
        __syncthreads();
    }
    if (i < n) excl[i] = sm[threadIdx.x] - v;      // exclusive within block
    if (threadIdx.x == 1023) bsum[blockIdx.x] = sm[1023];
}

__global__ void k_scan_tail(int* __restrict__ bsum, int nb) {
    if (threadIdx.x == 0 && blockIdx.x == 0) {
        int run = 0;
        for (int b = 0; b < nb; ++b) { int v = bsum[b]; bsum[b] = run; run += v; }
    }
}

__global__ void k_scan_add(int* __restrict__ excl, const int* __restrict__ bsum, int n) {
    int i = blockIdx.x * blockDim.x + threadIdx.x;
    if (i < n) excl[i] += bsum[i >> 10];
}

// bucket-fill CSR: nsrc/ews sorted by dst; ews = dinv[src]*dinv[dst]
__global__ void k_fill(const int* __restrict__ src, const int* __restrict__ dst,
                       const float* __restrict__ dinv, const int* __restrict__ rowptr,
                       int* __restrict__ cursor, int* __restrict__ nsrc,
                       float* __restrict__ ews, int E) {
    int e = blockIdx.x * blockDim.x + threadIdx.x;
    if (e >= E) return;
    int sv = src[e], dv = dst[e];
    int pos = rowptr[dv] + atomicAdd(&cursor[dv], 1);
    nsrc[pos] = sv;
    ews[pos]  = dinv[sv] * dinv[dv];
}

// ---------------------------------------------------------------------------
// Segmented GCN aggregation, one wave per dst node, lane owns 4 features.
//   acc = hW[v]*dinv[v]^2 + sum_{j in CSR[v]} hW[nsrc[j]] * ews[j]
//   y = relu(acc*scale + shift)   (bias+BN folded) -> hB (+ optional hF)
// No float atomics: pure gather + register accumulate.
// ---------------------------------------------------------------------------
__global__ void k_aggregate(const float* __restrict__ hW, const int* __restrict__ rowptr,
                            const int* __restrict__ nsrc, const float* __restrict__ ews,
                            const float* __restrict__ dinv, const float* __restrict__ scale,
                            const float* __restrict__ shift, float* __restrict__ hF,
                            __bf16* __restrict__ hB, int N, int E) {
    const int wid  = (int)((blockIdx.x * (long)blockDim.x + threadIdx.x) >> 5);
    const int lane = threadIdx.x & 31;
    if (wid >= N) return;
    const int f = lane << 2;

    const float di = dinv[wid];
    const float4 sl = *(const float4*)(hW + (long)wid * 128 + f);
    float a0 = sl.x * di * di, a1 = sl.y * di * di;
    float a2 = sl.z * di * di, a3 = sl.w * di * di;

    const int beg = rowptr[wid];
    const int end = (wid + 1 < N) ? rowptr[wid + 1] : E;
    for (int j = beg; j < end; ++j) {
        const int   s = nsrc[j];
        const float w = ews[j];
        const float4 m = *(const float4*)(hW + (long)s * 128 + f);
        a0 += m.x * w; a1 += m.y * w; a2 += m.z * w; a3 += m.w * w;
    }

    float y0 = a0 * scale[f + 0] + shift[f + 0]; y0 = y0 > 0.f ? y0 : 0.f;
    float y1 = a1 * scale[f + 1] + shift[f + 1]; y1 = y1 > 0.f ? y1 : 0.f;
    float y2 = a2 * scale[f + 2] + shift[f + 2]; y2 = y2 > 0.f ? y2 : 0.f;
    float y3 = a3 * scale[f + 3] + shift[f + 3]; y3 = y3 > 0.f ? y3 : 0.f;

    const long o = (long)wid * 128 + f;
    union { __bf16 b[4]; uint2 u; } pk;
    pk.b[0] = (__bf16)y0; pk.b[1] = (__bf16)y1; pk.b[2] = (__bf16)y2; pk.b[3] = (__bf16)y3;
    *(uint2*)(hB + o) = pk.u;
    if (hF) *(float4*)(hF + o) = make_float4(y0, y1, y2, y3);
}

// Wt[n*K+k] = bf16(W[k*N+n])
__global__ void k_transpose_bf16(const float* __restrict__ W, __bf16* __restrict__ Wt,
                                 int K, int N) {
    int i = blockIdx.x * blockDim.x + threadIdx.x;
    if (i >= K * N) return;
    int k = i / N, n = i % N;
    Wt[(long)n * K + k] = (__bf16)W[i];
}

// fold bias + BN(eval): scale = g*rsqrt(var+eps), shift = (bias-mean)*scale + beta
__global__ void k_bnfuse(const float* __restrict__ gamma, const float* __restrict__ beta,
                         const float* __restrict__ mean, const float* __restrict__ var,
                         const float* __restrict__ bias, float* __restrict__ scale,
                         float* __restrict__ shift, int dim) {
    int i = blockIdx.x * blockDim.x + threadIdx.x;
    if (i >= dim) return;
    float sc = gamma[i] * rsqrtf(var[i] + BN_EPS);
    scale[i] = sc;
    shift[i] = (bias[i] - mean[i]) * sc + beta[i];
}

// out[base+r, 0:2] = e2[r,:] @ W3 + b3   (K=128, N=2: VALU dot, v8bf loads)
__global__ void k_final(const __bf16* __restrict__ e2, const float* __restrict__ W3,
                        const float* __restrict__ b3, int rowbase, int rows,
                        float* __restrict__ out) {
    int r = blockIdx.x * blockDim.x + threadIdx.x;
    if (r >= rows) return;
    const v8bf* row = (const v8bf*)(e2 + (long)r * 128);
    float a0 = b3[0], a1 = b3[1];
#pragma unroll
    for (int c = 0; c < 16; ++c) {
        v8bf v = row[c];
#pragma unroll
        for (int t = 0; t < 8; ++t) {
            const int k = c * 8 + t;
            const float xv = (float)v[t];
            a0 += xv * W3[2 * k];
            a1 += xv * W3[2 * k + 1];
        }
    }
    long o = (long)(rowbase + r) * 2;
    out[o + 0] = a0;
    out[o + 1] = a1;
}

// ---------------------------------------------------------------------------

extern "C" void kernel_launch(void* const* d_in, const int* in_sizes, int n_in,
                              void* d_out, int out_size, void* d_ws, size_t ws_size,
                              hipStream_t stream) {
    (void)in_sizes; (void)n_in; (void)out_size; (void)ws_size;
    const int N = 100000, E = 800000, H = 128, NP = 500000;

    // Input leaves: top-level in setup_inputs() insertion order; nested params
    // flattened JAX-pytree style (dict keys sorted, lists in order).
    const float* x      = (const float*)d_in[0];
    const float* W[3]   = {(const float*)d_in[1], (const float*)d_in[7], (const float*)d_in[15]};
    const float* bb[3]  = {(const float*)d_in[2], (const float*)d_in[8], (const float*)d_in[16]};
    const float* beta[3]= {(const float*)d_in[3], (const float*)d_in[9], (const float*)d_in[17]};
    const float* gam[3] = {(const float*)d_in[4], (const float*)d_in[10], (const float*)d_in[18]};
    const float* mean[3]= {(const float*)d_in[5], (const float*)d_in[11], (const float*)d_in[19]};
    const float* var_[3]= {(const float*)d_in[6], (const float*)d_in[14], (const float*)d_in[22]};
    const float* sW[3]  = {nullptr, (const float*)d_in[12], (const float*)d_in[20]};
    const float* sb[3]  = {nullptr, (const float*)d_in[13], (const float*)d_in[21]};
    const float* mW1 = (const float*)d_in[23];   // [256,256]
    const float* mW2 = (const float*)d_in[24];   // [256,128]
    const float* mW3 = (const float*)d_in[25];   // [128,2]
    const float* mb1 = (const float*)d_in[26];
    const float* mb2 = (const float*)d_in[27];
    const float* mb3 = (const float*)d_in[28];
    const float* bn1b = (const float*)d_in[29], *bn1g = (const float*)d_in[30];
    const float* bn1m = (const float*)d_in[31], *bn1v = (const float*)d_in[32];
    const float* bn2b = (const float*)d_in[33], *bn2g = (const float*)d_in[34];
    const float* bn2m = (const float*)d_in[35], *bn2v = (const float*)d_in[36];
    const int* esrc = (const int*)d_in[37];
    const int* edst = esrc + E;
    const int* sArr = (const int*)d_in[38];
    const int* dArr = (const int*)d_in[39];
    float* out = (float*)d_out;

    // -------- workspace carve --------
    char* wsb = (char*)d_ws;
    size_t off = 0;
    auto carve = [&](size_t bytes) -> void* {
        void* p = wsb + off;
        off = (off + bytes + 255) & ~(size_t)255;
        return p;
    };
    float*  h     = (float*) carve((size_t)N * H * 4);
    __bf16* h_bf  = (__bf16*)carve((size_t)N * H * 2);
    float*  hW    = (float*) carve((size_t)N * H * 4);
    __bf16* hn_bf = (__bf16*)carve((size_t)N * H * 2);
    float*  dinv  = (float*) carve((size_t)N * 4);
    int*    cnt   = (int*)   carve((size_t)N * 4);
    int*    rowptr= (int*)   carve((size_t)N * 4);
    int*    bsum  = (int*)   carve(128 * 4);
    int*    cursor= (int*)   carve((size_t)N * 4);
    int*    nsrc  = (int*)   carve((size_t)E * 4);
    float*  ews   = (float*) carve((size_t)E * 4);
    __bf16* Wt[3], *SWt[3];
    for (int i = 0; i < 3; ++i) Wt[i] = (__bf16*)carve(128 * 128 * 2);
    SWt[1] = (__bf16*)carve(128 * 128 * 2);
    SWt[2] = (__bf16*)carve(128 * 128 * 2);
    __bf16* W1t = (__bf16*)carve(256 * 256 * 2);   // [256 cols x 256 k]
    __bf16* W2t = (__bf16*)carve(128 * 256 * 2);   // [128 cols x 256 k]
    float* scl[3], *shf[3];
    for (int i = 0; i < 3; ++i) { scl[i] = (float*)carve(128 * 4); shf[i] = (float*)carve(128 * 4); }
    float* sclM1 = (float*)carve(256 * 4); float* shfM1 = (float*)carve(256 * 4);
    float* sclM2 = (float*)carve(128 * 4); float* shfM2 = (float*)carve(128 * 4);
    const int CH = 65536;
    __bf16* e1 = (__bf16*)carve((size_t)CH * 256 * 2);
    __bf16* e2 = (__bf16*)carve((size_t)CH * 128 * 2);

    const long NH = (long)N * H;
    const int TB = 256;
    auto gb = [](long n, int b) { return (int)((n + b - 1) / b); };

    // -------- weight / BN prep --------
    for (int i = 0; i < 3; ++i)
        k_transpose_bf16<<<gb(128 * 128, TB), TB, 0, stream>>>(W[i], Wt[i], 128, 128);
    k_transpose_bf16<<<gb(128 * 128, TB), TB, 0, stream>>>(sW[1], SWt[1], 128, 128);
    k_transpose_bf16<<<gb(128 * 128, TB), TB, 0, stream>>>(sW[2], SWt[2], 128, 128);
    k_transpose_bf16<<<gb(256 * 256, TB), TB, 0, stream>>>(mW1, W1t, 256, 256);
    k_transpose_bf16<<<gb(256 * 128, TB), TB, 0, stream>>>(mW2, W2t, 256, 128);
    for (int i = 0; i < 3; ++i)
        k_bnfuse<<<1, 128, 0, stream>>>(gam[i], beta[i], mean[i], var_[i], bb[i], scl[i], shf[i], 128);
    k_bnfuse<<<1, 256, 0, stream>>>(bn1g, bn1b, bn1m, bn1v, mb1, sclM1, shfM1, 256);
    k_bnfuse<<<1, 128, 0, stream>>>(bn2g, bn2b, bn2m, bn2v, mb2, sclM2, shfM2, 128);

    // -------- CSR-by-dst build (atomic-free aggregation; reused 3 layers) ----
    const int nb = (N + 1023) / 1024;
    k_set_i32<<<gb(N, TB), TB, 0, stream>>>(cnt, 0, N);
    k_hist<<<gb(E, TB), TB, 0, stream>>>(edst, cnt, E);
    k_dinv<<<gb(N, TB), TB, 0, stream>>>(cnt, dinv, N);
    k_scan_block<<<nb, 1024, 0, stream>>>(cnt, rowptr, bsum, N);
    k_scan_tail<<<1, 1, 0, stream>>>(bsum, nb);
    k_scan_add<<<gb(N, TB), TB, 0, stream>>>(rowptr, bsum, N);
    k_set_i32<<<gb(N, TB), TB, 0, stream>>>(cursor, 0, N);
    k_fill<<<gb(E, TB), TB, 0, stream>>>(esrc, edst, dinv, rowptr, cursor, nsrc, ews, E);

    // -------- GCN layers --------
    k_f32_to_bf16<<<gb(NH, TB), TB, 0, stream>>>(x, h_bf, NH);   // layer-0 A operand
    const int gemmBlocks = (N + 63) / 64;
    const int aggBlocks  = gb((long)N * 32, TB);
    for (int i = 0; i < 3; ++i) {
        // hW = h @ W    (bf16 WMMA, f32 accum)
        gemm_wmma<128, 128, 0, false><<<gemmBlocks, 128, 0, stream>>>(
            h_bf, Wt[i], nullptr, nullptr, nullptr, hW, nullptr, N,
            nullptr, nullptr, nullptr);
        // hn = relu(bn(selfloop + segmented-sum + b)) — wave-per-node gather
        if (i == 0) {
            k_aggregate<<<aggBlocks, TB, 0, stream>>>(hW, rowptr, nsrc, ews, dinv,
                                                      scl[i], shf[i], h, h_bf, N, E);
        } else {
            k_aggregate<<<aggBlocks, TB, 0, stream>>>(hW, rowptr, nsrc, ews, dinv,
                                                      scl[i], shf[i], nullptr, hn_bf, N, E);
            // h = h + hn @ skip_W + skip_b   (also refresh h_bf)
            gemm_wmma<128, 128, 1, false><<<gemmBlocks, 128, 0, stream>>>(
                hn_bf, SWt[i], nullptr, sb[i], h, h, h_bf, N,
                nullptr, nullptr, nullptr);
        }
    }

    // -------- edge-pair MLP, chunked over NP --------
    for (int base = 0; base < NP; base += CH) {
        const int cs = (NP - base) < CH ? (NP - base) : CH;
        const int blks = (cs + 63) / 64;
        // e1 = relu(bn1(concat(h[s],h[d]) @ W1 + b1)) — gather fused into A frags
        gemm_wmma<256, 256, 2, true><<<blks, 128, 0, stream>>>(
            nullptr, W1t, sclM1, shfM1, nullptr, nullptr, e1, cs,
            sArr + base, dArr + base, h_bf);
        // e2 = relu(bn2(e1 @ W2 + b2))
        gemm_wmma<256, 128, 2, false><<<blks, 128, 0, stream>>>(
            e1, W2t, sclM2, shfM2, nullptr, nullptr, e2, cs,
            nullptr, nullptr, nullptr);
        // out = e2 @ W3 + b3
        k_final<<<gb(cs, TB), TB, 0, stream>>>(e2, mW3, mb3, base, cs, out);
    }
}
